// StructureEncoder_64115271795146
// MI455X (gfx1250) — matrix operand order repaired and checked by reading, not compiled, not run on hardware
//
#include <hip/hip_runtime.h>
#include <hip/hip_bf16.h>

// ---------------------------------------------------------------------------
// GATv2 structure encoder for gfx1250 (MI455X).
// Dense GEMMs: v_wmma_f32_16x16x32_bf16, A-tile staged via TDM
// (tensor_load_to_lds + s_wait_tensorcnt). Edge phase: L2-resident gathers +
// float atomics (xl/xr are 20.5MB each -> fully L2-cached on 192MB L2).
// ---------------------------------------------------------------------------

#define NN   20000
#define EE   640000
#define ETOT (EE + NN)
#define KIN  128
#define HIDD 256
#define NH   8
#define CH   32
#define NL   5
#define NG   64
#define NEG_SLOPE 0.2f

#define USE_TDM 1   // tensor_load_to_lds staging of the A tile

typedef __attribute__((ext_vector_type(16))) __bf16 v16bf;
typedef __attribute__((ext_vector_type(8)))  float  v8f;
typedef __attribute__((ext_vector_type(8)))  unsigned int v8u;
typedef __attribute__((ext_vector_type(4)))  unsigned int u32x4;

__device__ __forceinline__ unsigned short f32_to_bf16_rne(float f) {
    unsigned int u = __float_as_uint(f);
    unsigned int r = (u + 0x7FFFu + ((u >> 16) & 1u)) >> 16;
    return (unsigned short)r;
}

__device__ __forceinline__ void atomicMaxFloat(float* addr, float val) {
    if (val >= 0.0f) atomicMax((int*)addr, __float_as_int(val));
    else             atomicMin((unsigned int*)addr, __float_as_uint(val));
}

// ---------------------------------------------------------------------------
// Utility kernels
// ---------------------------------------------------------------------------
__global__ void fill_f32_kernel(float* p, float v, int n) {
    int i = blockIdx.x * blockDim.x + threadIdx.x;
    if (i < n) p[i] = v;
}

__global__ void cvt_f32_bf16_kernel(const float* in, unsigned short* out, int n) {
    int i = blockIdx.x * blockDim.x + threadIdx.x;
    if (i < n) out[i] = f32_to_bf16_rne(in[i]);
}

// ---------------------------------------------------------------------------
// WMMA GEMM (dual weight-set): for by<4 computes C0 = A@B0 + bias0 over cols
// [by*64, by*64+64); for by>=4 same with set 1. Block: 256 threads (8 waves),
// block tile 128x64, wave tile 32x32 (2x2 WMMA 16x16x32 bf16 tiles).
// A tile (128x32, row-major) staged by the Tensor Data Mover; B tile staged
// transposed (Bs[n][k]) so fragment loads are packed-2xbf16 dword reads.
// ---------------------------------------------------------------------------
__global__ __launch_bounds__(256)
void gemm_bf16_wmma_kernel(const unsigned short* __restrict__ A,
                           const unsigned short* __restrict__ B0,
                           const float* __restrict__ bias0,
                           float* __restrict__ C0,
                           unsigned short* __restrict__ Cbf0,
                           const unsigned short* __restrict__ B1,
                           const float* __restrict__ bias1,
                           float* __restrict__ C1,
                           unsigned short* __restrict__ Cbf1,
                           int M, int K) {
    __shared__ __align__(16) unsigned short As[128 * 32];
    __shared__ __align__(16) unsigned short Bs[64 * 32];

    const int tid  = threadIdx.x;
    const int wave = tid >> 5;
    const int lane = tid & 31;
    const int half = lane >> 4;
    const int mlo  = lane & 15;
    const int wm   = wave & 3;
    const int wn   = wave >> 2;
    const int bm0  = blockIdx.x * 128;
    const int by   = blockIdx.y;
    const int bn0  = (by & 3) * 64;

    const unsigned short* B    = (by < 4) ? B0 : B1;
    const float*          bias = (by < 4) ? bias0 : bias1;
    float*                C    = (by < 4) ? C0 : C1;
    unsigned short*       Cbf  = (by < 4) ? Cbf0 : Cbf1;

#if USE_TDM
    const unsigned lds_As  = (unsigned)(unsigned long long)(void*)As; // flat addr[31:0] == LDS offset
    const unsigned rows_rem = (unsigned)(M - bm0);
#endif

    v8f acc[2][2] = {};

    for (int kb = 0; kb < K; kb += 32) {
        __syncthreads();
#if USE_TDM
        // ---- stage A tile (128 x 32 bf16, row-major) via Tensor Data Mover ----
        if (wave == 0) {
            unsigned long long gaddr =
                (unsigned long long)(A + (size_t)bm0 * K + kb);
            u32x4 g0;
            g0[0] = 1u;                                   // count=1 valid descriptor
            g0[1] = lds_As;                               // lds_addr
            g0[2] = (unsigned)gaddr;                      // global_addr[31:0]
            g0[3] = (unsigned)((gaddr >> 32) & 0x1FFFFFFull) | (2u << 30); // addr[56:32] | type=2
            v8u g1;
            unsigned kd = (unsigned)K;
            g1[0] = (1u << 16);                           // data_size = 2B
            g1[1] = (kd & 0xFFFFu) << 16;                 // tensor_dim0[15:0]
            g1[2] = ((kd >> 16) & 0xFFFFu) |              // tensor_dim0[31:16]
                    ((rows_rem & 0xFFFFu) << 16);         // tensor_dim1[15:0]
            g1[3] = ((rows_rem >> 16) & 0xFFFFu) |        // tensor_dim1[31:16]
                    (32u << 16);                          // tile_dim0 = 32
            g1[4] = 128u;                                 // tile_dim1 = 128, tile_dim2 = 0
            g1[5] = kd;                                   // tensor_dim0_stride[31:0] = K
            g1[6] = 0u;                                   // stride0 hi / stride1 lo
            g1[7] = 0u;
            asm volatile("tensor_load_to_lds %0, %1" :: "s"(g0), "s"(g1) : "memory");
            __builtin_amdgcn_s_wait_tensorcnt(0);
        }
#else
        {   // fallback: per-lane staging of A tile
            int row  = tid >> 1;
            int col0 = (tid & 1) * 16;
            int grow = bm0 + row;
            uint4 z = make_uint4(0u, 0u, 0u, 0u);
            uint4 d0 = z, d1 = z;
            if (grow < M) {
                const uint4* src = (const uint4*)(A + (size_t)grow * K + kb + col0);
                d0 = src[0];
                d1 = src[1];
            }
            uint4* dst = (uint4*)(As + row * 32 + col0);
            dst[0] = d0;
            dst[1] = d1;
        }
#endif
        // ---- stage B tile transposed: Bs[n][k], n in [0,64), k in [0,32) ----
        {
            int k  = tid >> 3;
            int n0 = (tid & 7) * 8;
            const uint4* src = (const uint4*)(B + (size_t)(kb + k) * HIDD + bn0 + n0);
            uint4 d = src[0];
            unsigned short e[8];
            e[0] = (unsigned short)(d.x & 0xFFFF); e[1] = (unsigned short)(d.x >> 16);
            e[2] = (unsigned short)(d.y & 0xFFFF); e[3] = (unsigned short)(d.y >> 16);
            e[4] = (unsigned short)(d.z & 0xFFFF); e[5] = (unsigned short)(d.z >> 16);
            e[6] = (unsigned short)(d.w & 0xFFFF); e[7] = (unsigned short)(d.w >> 16);
#pragma unroll
            for (int j = 0; j < 8; ++j) Bs[(n0 + j) * 32 + k] = e[j];
        }
        __syncthreads();

        // ---- fragments ----
        v16bf fa[2], fb[2];
#pragma unroll
        for (int r = 0; r < 2; ++r) {
            int arow = wm * 32 + r * 16 + mlo;
            const unsigned int* Ap = (const unsigned int*)(As + arow * 32);
            v8u pa;
#pragma unroll
            for (int v = 0; v < 8; ++v) {
                // VGPR v holds K = (v>>2)*16 + half*8 + (v&3)*2 , K+1
                int kk = (((v >> 2) * 16) + half * 8 + ((v & 3) * 2)) >> 1;
                pa[v] = Ap[kk];
            }
            fa[r] = __builtin_bit_cast(v16bf, pa);
        }
#pragma unroll
        for (int s = 0; s < 2; ++s) {
            int bcol = wn * 32 + s * 16 + mlo;
            const unsigned int* Bp = (const unsigned int*)(Bs + bcol * 32);
            v8u pb;
#pragma unroll
            for (int v = 0; v < 8; ++v) {
                int kk = (v >> 2) * 8 + half * 4 + (v & 3);   // K = (v>>2)*16 + half*8 + (v&3)*2 , +1
                pb[v] = Bp[kk];
            }
            fb[s] = __builtin_bit_cast(v16bf, pb);
        }
#pragma unroll
        for (int r = 0; r < 2; ++r)
#pragma unroll
            for (int s = 0; s < 2; ++s)
                acc[r][s] = __builtin_amdgcn_wmma_f32_16x16x32_bf16(
                    false, fa[r], false, fb[s], (short)0, acc[r][s], false, false);
    }

    // ---- store: D VGPR v -> row v + 8*half, col = mlo ----
#pragma unroll
    for (int r = 0; r < 2; ++r) {
#pragma unroll
        for (int s = 0; s < 2; ++s) {
            int gcol = bn0 + wn * 32 + s * 16 + mlo;
            float bv = bias[gcol];
#pragma unroll
            for (int v = 0; v < 8; ++v) {
                int grow = bm0 + wm * 32 + r * 16 + v + 8 * half;
                if (grow < M) {
                    float val = acc[r][s][v] + bv;
                    size_t idx = (size_t)grow * HIDD + gcol;
                    C[idx] = val;
                    if (Cbf) Cbf[idx] = f32_to_bf16_rne(val);
                }
            }
        }
    }
}

// ---------------------------------------------------------------------------
// Edge phase kernels: one thread per (edge, head)
// ---------------------------------------------------------------------------
__device__ __forceinline__ void edge_endpoints(const long long* ei, int e,
                                               int& src, int& dst) {
    if (e < EE) { src = (int)ei[e]; dst = (int)ei[EE + e]; }
    else        { src = e - EE; dst = e - EE; }
}

__global__ void edge_logits_kernel(const long long* __restrict__ ei,
                                   const float* __restrict__ xl,
                                   const float* __restrict__ xr,
                                   const float* __restrict__ att,   // [NH,CH] for this layer
                                   float* __restrict__ logits,
                                   float* __restrict__ mbuf) {
    int t = blockIdx.x * blockDim.x + threadIdx.x;
    if (t >= ETOT * NH) return;
    int e = t >> 3, h = t & 7;
    int src, dst;
    edge_endpoints(ei, e, src, dst);
    const float4* pl = (const float4*)(xl + (size_t)src * HIDD + h * CH);
    const float4* pr = (const float4*)(xr + (size_t)dst * HIDD + h * CH);
    const float4* pw = (const float4*)(att + h * CH);
    float acc = 0.0f;
#pragma unroll
    for (int j = 0; j < 8; ++j) {
        float4 a = pl[j], b = pr[j], w = pw[j];
        float v;
        v = a.x + b.x; v = v > 0.0f ? v : NEG_SLOPE * v; acc += v * w.x;
        v = a.y + b.y; v = v > 0.0f ? v : NEG_SLOPE * v; acc += v * w.y;
        v = a.z + b.z; v = v > 0.0f ? v : NEG_SLOPE * v; acc += v * w.z;
        v = a.w + b.w; v = v > 0.0f ? v : NEG_SLOPE * v; acc += v * w.w;
    }
    logits[t] = acc;
    atomicMaxFloat(&mbuf[dst * NH + h], acc);
}

__global__ void edge_exp_sum_kernel(const long long* __restrict__ ei,
                                    float* __restrict__ logits,
                                    const float* __restrict__ mbuf,
                                    float* __restrict__ sbuf) {
    int t = blockIdx.x * blockDim.x + threadIdx.x;
    if (t >= ETOT * NH) return;
    int e = t >> 3, h = t & 7;
    int src, dst;
    edge_endpoints(ei, e, src, dst);
    float a = __expf(logits[t] - mbuf[dst * NH + h]);
    logits[t] = a;
    atomicAdd(&sbuf[dst * NH + h], a);
}

__global__ void edge_aggregate_kernel(const long long* __restrict__ ei,
                                      const float* __restrict__ logits,
                                      const float* __restrict__ sbuf,
                                      const float* __restrict__ xl,
                                      float* __restrict__ conv) {
    int t = blockIdx.x * blockDim.x + threadIdx.x;
    if (t >= ETOT * NH) return;
    int e = t >> 3, h = t & 7;
    int src, dst;
    edge_endpoints(ei, e, src, dst);
    float alpha = logits[t] / (sbuf[dst * NH + h] + 1e-16f);
    const float* xp = xl + (size_t)src * HIDD + h * CH;
    float* op = conv + (size_t)dst * HIDD + h * CH;
#pragma unroll
    for (int c = 0; c < CH; ++c) atomicAdd(op + c, alpha * xp[c]);
}

// ---------------------------------------------------------------------------
// Residual + ELU + LayerNorm, fused; one block (256 threads) per node.
// Consumes conv and re-zeroes it for the next layer's accumulation; refreshes
// the bf16 copy of h for the next layer's WMMA GEMM.
// ---------------------------------------------------------------------------
__global__ __launch_bounds__(256)
void post_layer_kernel(float* __restrict__ h,
                       float* __restrict__ conv,
                       const float* __restrict__ out_bias,
                       const float* __restrict__ ln_g,
                       const float* __restrict__ ln_b,
                       unsigned short* __restrict__ hbf) {
    __shared__ float red[256];
    int node = blockIdx.x;
    int c = threadIdx.x;
    size_t idx = (size_t)node * HIDD + c;
    float hv = h[idx];
    float cv = conv[idx] + out_bias[c];
    conv[idx] = 0.0f;   // ready for next layer's atomic accumulation
    float el = cv > 0.0f ? cv : (__expf(cv) - 1.0f);
    float v = hv + el;

    red[c] = v; __syncthreads();
    for (int o = 128; o > 0; o >>= 1) { if (c < o) red[c] += red[c + o]; __syncthreads(); }
    float mu = red[0] / (float)HIDD; __syncthreads();

    float d = v - mu;
    red[c] = d * d; __syncthreads();
    for (int o = 128; o > 0; o >>= 1) { if (c < o) red[c] += red[c + o]; __syncthreads(); }
    float var = red[0] / (float)HIDD;

    float y = d * rsqrtf(var + 1e-5f) * ln_g[c] + ln_b[c];
    h[idx] = y;
    hbf[idx] = f32_to_bf16_rne(y);
}

// ---------------------------------------------------------------------------
// Global mean pool + batch copy
// ---------------------------------------------------------------------------
__global__ void pool_accum_kernel(const float* __restrict__ h,
                                  const long long* __restrict__ batch,
                                  float* __restrict__ gsum,
                                  float* __restrict__ counts) {
    int t = blockIdx.x * blockDim.x + threadIdx.x;
    if (t >= NN * HIDD) return;
    int node = t >> 8, c = t & 255;
    int g = (int)batch[node];
    atomicAdd(&gsum[g * HIDD + c], h[t]);
    if (c == 0) atomicAdd(&counts[g], 1.0f);
}

__global__ void pool_div_kernel(float* __restrict__ gsum,
                                const float* __restrict__ counts) {
    int t = blockIdx.x * blockDim.x + threadIdx.x;
    if (t >= NG * HIDD) return;
    gsum[t] /= fmaxf(counts[t >> 8], 1.0f);
}

__global__ void batch_copy_kernel(const long long* __restrict__ batch,
                                  float* __restrict__ out) {
    int i = blockIdx.x * blockDim.x + threadIdx.x;
    if (i < NN) out[i] = (float)batch[i];
}

// ---------------------------------------------------------------------------
// Host orchestration
// ---------------------------------------------------------------------------
extern "C" void kernel_launch(void* const* d_in, const int* in_sizes, int n_in,
                              void* d_out, int out_size, void* d_ws, size_t ws_size,
                              hipStream_t stream) {
    const float*     x        = (const float*)d_in[0];
    const long long* ei       = (const long long*)d_in[1];
    const long long* batch    = (const long long*)d_in[2];
    const float*     proj_W   = (const float*)d_in[3];
    const float*     proj_b   = (const float*)d_in[4];
    const float*     Wl       = (const float*)d_in[5];
    const float*     bl       = (const float*)d_in[6];
    const float*     Wr       = (const float*)d_in[7];
    const float*     br       = (const float*)d_in[8];
    const float*     att      = (const float*)d_in[9];
    const float*     out_bias = (const float*)d_in[10];
    const float*     ln_g     = (const float*)d_in[11];
    const float*     ln_b     = (const float*)d_in[12];

    // d_out layout: graph_emb [NG*HIDD] | h [NN*HIDD] | batch [NN]
    float* gsum  = (float*)d_out;
    float* h     = gsum + NG * HIDD;
    float* bout  = h + (size_t)NN * HIDD;

    // workspace carve-out (256B aligned)
    char* wsp = (char*)d_ws;
    auto carve = [&](size_t bytes) -> char* {
        char* p = wsp;
        wsp += (bytes + 255) & ~(size_t)255;
        return p;
    };
    unsigned short* hbf    = (unsigned short*)carve((size_t)NN * HIDD * 2);
    unsigned short* xbf    = (unsigned short*)carve((size_t)NN * KIN * 2);
    unsigned short* pWbf   = (unsigned short*)carve((size_t)KIN * HIDD * 2);
    unsigned short* Wlbf   = (unsigned short*)carve((size_t)NL * HIDD * HIDD * 2);
    unsigned short* Wrbf   = (unsigned short*)carve((size_t)NL * HIDD * HIDD * 2);
    float*          xl     = (float*)carve((size_t)NN * HIDD * 4);
    float*          xr     = (float*)carve((size_t)NN * HIDD * 4);
    float*          conv   = (float*)carve((size_t)NN * HIDD * 4);
    float*          logits = (float*)carve((size_t)ETOT * NH * 4);
    float*          mbuf   = (float*)carve((size_t)NN * NH * 4);
    float*          sbuf   = (float*)carve((size_t)NN * NH * 4);
    float*          counts = (float*)carve((size_t)NG * 4);

    const int T = 256;
    auto gs = [](long long n) { return (unsigned)((n + 255) / 256); };

    // weight/input conversions to bf16
    cvt_f32_bf16_kernel<<<gs(KIN * HIDD), T, 0, stream>>>(proj_W, pWbf, KIN * HIDD);
    cvt_f32_bf16_kernel<<<gs((long long)NL * HIDD * HIDD), T, 0, stream>>>(Wl, Wlbf, NL * HIDD * HIDD);
    cvt_f32_bf16_kernel<<<gs((long long)NL * HIDD * HIDD), T, 0, stream>>>(Wr, Wrbf, NL * HIDD * HIDD);
    cvt_f32_bf16_kernel<<<gs((long long)NN * KIN), T, 0, stream>>>(x, xbf, NN * KIN);

    dim3 ggrid1((NN + 127) / 128, 4);   // single weight set (projection)
    dim3 ggrid2((NN + 127) / 128, 8);   // dual weight set (Wl + Wr)

    // projection: h = x @ proj_W + proj_b  (also produce bf16 copy)
    gemm_bf16_wmma_kernel<<<ggrid1, T, 0, stream>>>(xbf, pWbf, proj_b, h, hbf,
                                                    pWbf, proj_b, h, hbf, NN, KIN);

    // conv starts zeroed; post_layer re-zeroes it each layer
    fill_f32_kernel<<<gs((long long)NN * HIDD), T, 0, stream>>>(conv, 0.0f, NN * HIDD);

    const long long nEH = (long long)ETOT * NH;
    for (int i = 0; i < NL; ++i) {
        const unsigned short* Wli = Wlbf + (size_t)i * HIDD * HIDD;
        const unsigned short* Wri = Wrbf + (size_t)i * HIDD * HIDD;

        // xl = h @ Wl + bl ; xr = h @ Wr + br   (single fused launch)
        gemm_bf16_wmma_kernel<<<ggrid2, T, 0, stream>>>(hbf,
                                                        Wli, bl + i * HIDD, xl, nullptr,
                                                        Wri, br + i * HIDD, xr, nullptr,
                                                        NN, HIDD);

        fill_f32_kernel<<<gs(NN * NH), T, 0, stream>>>(mbuf, -__builtin_huge_valf(), NN * NH);
        fill_f32_kernel<<<gs(NN * NH), T, 0, stream>>>(sbuf, 0.0f, NN * NH);

        edge_logits_kernel<<<gs(nEH), T, 0, stream>>>(ei, xl, xr, att + i * NH * CH, logits, mbuf);
        edge_exp_sum_kernel<<<gs(nEH), T, 0, stream>>>(ei, logits, mbuf, sbuf);
        edge_aggregate_kernel<<<gs(nEH), T, 0, stream>>>(ei, logits, sbuf, xl, conv);

        post_layer_kernel<<<NN, T, 0, stream>>>(h, conv, out_bias + i * HIDD,
                                                ln_g + i * HIDD, ln_b + i * HIDD, hbf);
    }

    // global mean pool
    fill_f32_kernel<<<gs(NG * HIDD), T, 0, stream>>>(gsum, 0.0f, NG * HIDD);
    fill_f32_kernel<<<1, T, 0, stream>>>(counts, 0.0f, NG);
    pool_accum_kernel<<<gs((long long)NN * HIDD), T, 0, stream>>>(h, batch, gsum, counts);
    pool_div_kernel<<<gs(NG * HIDD), T, 0, stream>>>(gsum, counts);
    batch_copy_kernel<<<gs(NN), T, 0, stream>>>(batch, bout);
}
